// UnSupLoss_57561151701206
// MI455X (gfx1250) — matrix operand order repaired and checked by reading, not compiled, not run on hardware
//
#include <hip/hip_runtime.h>
#include <hip/hip_bf16.h>
#include <math.h>

// ---------------------------------------------------------------------------
// UnSupLoss on MI455X (gfx1250): fused normalized-GEMM + log-softmax-CE.
// sim = z z^T via split-f16 WMMA (z_hi + z_lo), f32 accumulate:
//   sim ~= Ahi*Bhi + Ahi*Blo + Alo*Bhi   (~1e-7 relative, fp32-class)
// Block = 8 waves x 16 rows = 128 rows; all waves sweep the SAME column
// tiles so B loads hit the shared WGP$ (8x L2 traffic reduction).
// 3 independent WMMA accumulator chains break the RAW serialization.
// Column dim split into 4 slices for full-GPU parallelism (256 blocks).
// ---------------------------------------------------------------------------

typedef __attribute__((ext_vector_type(16))) _Float16 v16h;
typedef __attribute__((ext_vector_type(8)))  _Float16 v8h;
typedef __attribute__((ext_vector_type(8)))  float    v8f;

#define NROWS 8192
#define HDIM  768
#define KCHUNKS (HDIM / 32)     // 24
#define NSLICES 4
#define TILES_PER_SLICE (NROWS / 16 / NSLICES)  // 128
#define TEMP_INV 20.0f          // 1/0.05
#define LSE_SHIFT 20.0f

// ---------------- Pass 1: row L2-normalize + split into f16 hi/lo ----------
__global__ void __launch_bounds__(256)
normalize_split_kernel(const float* __restrict__ x,
                       _Float16* __restrict__ zhi,
                       _Float16* __restrict__ zlo) {
  __shared__ float red[256];
  const int row = blockIdx.x;
  const int t   = threadIdx.x;
  const float* xr = x + (size_t)row * HDIM;
  float v0 = xr[t], v1 = xr[t + 256], v2 = xr[t + 512];
  red[t] = v0 * v0 + v1 * v1 + v2 * v2;
  __syncthreads();
  for (int s = 128; s > 0; s >>= 1) {
    if (t < s) red[t] += red[t + s];
    __syncthreads();
  }
  const float inv = 1.0f / fmaxf(sqrtf(red[0]), 1e-8f);
  const size_t base = (size_t)row * HDIM;
  float z0 = v0 * inv, z1 = v1 * inv, z2 = v2 * inv;
  _Float16 h0 = (_Float16)z0, h1 = (_Float16)z1, h2 = (_Float16)z2;
  zhi[base + t]       = h0;
  zhi[base + t + 256] = h1;
  zhi[base + t + 512] = h2;
  zlo[base + t]       = (_Float16)(z0 - (float)h0);
  zlo[base + t + 256] = (_Float16)(z1 - (float)h1);
  zlo[base + t + 512] = (_Float16)(z2 - (float)h2);
}

// ---------------- WMMA fragment loaders (contiguous k order, same in A and B) -----------
// A (16x32, f16): lane<16 -> K=k0..k0+15 contiguous
//                 lane>=16 -> K=k0+16..k0+31 contiguous
__device__ __forceinline__ v16h loadA(const _Float16* __restrict__ z,
                                      int rowBase, int k0, int lane) {
  const int m  = rowBase + (lane & 15);
  const int kb = k0 + ((lane & 16) ? 16 : 0);
  const _Float16* p = z + (size_t)m * HDIM + kb;
  v8h a = *reinterpret_cast<const v8h*>(p);
  v8h b = *reinterpret_cast<const v8h*>(p + 8);
  v16h r;
#pragma unroll
  for (int i = 0; i < 8; ++i) { r[i] = a[i]; r[i + 8] = b[i]; }
  return r;
}

// B (32x16, f16): lane<16 (N=lane) holds K=k0..k0+15 contiguous;
//                 lane>=16 holds K=k0+16..k0+31 contiguous.
__device__ __forceinline__ v16h loadB(const _Float16* __restrict__ z,
                                      int colBase, int k0, int lane) {
  const int n  = colBase + (lane & 15);
  const int kb = k0 + ((lane & 16) ? 16 : 0);
  return *reinterpret_cast<const v16h*>(z + (size_t)n * HDIM + kb);
}

// ---------------- Pass 2: fused sim-tile GEMM + streaming logsumexp --------
// grid = (NSLICES, 64). Block: wave w owns rows [rowBlk*128 + w*16, +16).
// All 8 waves iterate identical column tiles -> B served from WGP$.
__global__ void __launch_bounds__(256, 2)
sim_lse_kernel(const _Float16* __restrict__ zhi,
               const _Float16* __restrict__ zlo,
               float* __restrict__ sums,     // [NSLICES][NROWS]
               float* __restrict__ pairv) {  // [NROWS]
  const int slice  = blockIdx.x;          // 0..3
  const int rowBlk = blockIdx.y;          // 0..63
  const int wave   = threadIdx.x >> 5;
  const int lane   = threadIdx.x & 31;
  const int rowBase = rowBlk * 128 + wave * 16;
  const int nloc  = lane & 15;
  const int hhalf = lane >> 4;            // 0: rows 0..7, 1: rows 8..15

  // Preload this wave's A fragments (hi+lo) for all K: 384 VGPRs, loop-invariant.
  v16h ahf[KCHUNKS], alf[KCHUNKS];
#pragma unroll
  for (int kk = 0; kk < KCHUNKS; ++kk) {
    ahf[kk] = loadA(zhi, rowBase, kk * 32, lane);
    alf[kk] = loadA(zlo, rowBase, kk * 32, lane);
  }

  float sumexp[8];
#pragma unroll
  for (int j = 0; j < 8; ++j) sumexp[j] = 0.0f;

  const int ct0 = slice * TILES_PER_SLICE;
  for (int ct = ct0; ct < ct0 + TILES_PER_SLICE; ++ct) {
    const int colBase = ct * 16;
    v8f acc0 = {}, acc1 = {}, acc2 = {};   // 3 independent WMMA chains
#pragma unroll
    for (int kk = 0; kk < KCHUNKS; ++kk) {
      v16h bh = loadB(zhi, colBase, kk * 32, lane);
      v16h bl = loadB(zlo, colBase, kk * 32, lane);
      acc0 = __builtin_amdgcn_wmma_f32_16x16x32_f16(false, ahf[kk], false, bh,
                                                    (short)0, acc0, false, false);
      acc1 = __builtin_amdgcn_wmma_f32_16x16x32_f16(false, ahf[kk], false, bl,
                                                    (short)0, acc1, false, false);
      acc2 = __builtin_amdgcn_wmma_f32_16x16x32_f16(false, alf[kk], false, bh,
                                                    (short)0, acc2, false, false);
    }
    const int nglob = colBase + nloc;
#pragma unroll
    for (int j = 0; j < 8; ++j) {
      const int mglob = rowBase + j + hhalf * 8;
      float logit = (acc0[j] + acc1[j] + acc2[j]) * TEMP_INV;
      if (nglob == (mglob ^ 1)) pairv[mglob] = logit;   // target logit (1 writer)
      if (nglob == mglob) logit = -__builtin_inff();    // diagonal mask
      sumexp[j] += expf(logit - LSE_SHIFT);
    }
  }

  // Butterfly-reduce across the 16 lanes of each half (halves hold disjoint rows).
#pragma unroll
  for (int j = 0; j < 8; ++j) {
    float s = sumexp[j];
    s += __shfl_xor(s, 1, 32);
    s += __shfl_xor(s, 2, 32);
    s += __shfl_xor(s, 4, 32);
    s += __shfl_xor(s, 8, 32);
    sumexp[j] = s;
  }
  if (nloc == 0) {  // lanes 0 (rows 0..7) and 16 (rows 8..15)
#pragma unroll
    for (int j = 0; j < 8; ++j)
      sums[(size_t)slice * NROWS + rowBase + hhalf * 8 + j] = sumexp[j];
  }
}

// ---------------- Pass 3: combine slices, final deterministic reduction ----
__global__ void __launch_bounds__(256)
finalize_kernel(const float* __restrict__ sums,   // [NSLICES][NROWS]
                const float* __restrict__ pairv,  // [NROWS]
                float* __restrict__ out) {
  __shared__ float red[256];
  float local = 0.0f;
  for (int i = threadIdx.x; i < NROWS; i += 256) {
    float s = sums[i] + sums[NROWS + i] + sums[2 * NROWS + i] + sums[3 * NROWS + i];
    local += (LSE_SHIFT + logf(s)) - pairv[i];
  }
  red[threadIdx.x] = local;
  __syncthreads();
  for (int s = 128; s > 0; s >>= 1) {
    if (threadIdx.x < s) red[threadIdx.x] += red[threadIdx.x + s];
    __syncthreads();
  }
  if (threadIdx.x == 0) out[0] = red[0] / (float)NROWS;
}

// ---------------------------------------------------------------------------
extern "C" void kernel_launch(void* const* d_in, const int* in_sizes, int n_in,
                              void* d_out, int out_size, void* d_ws, size_t ws_size,
                              hipStream_t stream) {
  (void)in_sizes; (void)n_in; (void)out_size; (void)ws_size;
  const float* x = (const float*)d_in[0];
  float* out = (float*)d_out;

  _Float16* zhi = (_Float16*)d_ws;                         // 12.58 MB
  _Float16* zlo = zhi + (size_t)NROWS * HDIM;              // 12.58 MB
  float* sums  = (float*)(zlo + (size_t)NROWS * HDIM);     // 128 KB
  float* pairv = sums + (size_t)NSLICES * NROWS;           // 32 KB

  normalize_split_kernel<<<NROWS, 256, 0, stream>>>(x, zhi, zlo);
  sim_lse_kernel<<<dim3(NSLICES, NROWS / 128), 256, 0, stream>>>(zhi, zlo, sums, pairv);
  finalize_kernel<<<1, 256, 0, stream>>>(sums, pairv, out);
}
